// BAAMamba_7224134992193
// MI455X (gfx1250) — matrix operand order repaired and checked by reading, not compiled, not run on hardware
//
#include <hip/hip_runtime.h>
#include <hip/hip_bf16.h>

// ---------------- model constants ----------------
#define DIRS    4
#define BATCH   8
#define DEPTH   8
#define DCH     192     // D (embed dim)
#define LSEQ    196
#define HGRID   14
#define DIN     384     // inner dim
#define DSTATE  16
#define DCONV   4
#define DTR     12
#define NCLS    1000
#define MROWS   (BATCH*LSEQ)   // 1568 (49 tiles of 32)
#define MG      (DIRS*MROWS)   // 6272
#define KPATCH  768            // 3*16*16
#define XZW     (2*DIN)        // 768
#define LN_EPS  1e-5f

#define KSUB    96             // K superchunk (3 WMMA k-steps)
#define AROWPAD 104            // LDS row stride for A tile (bank spread)

typedef __attribute__((ext_vector_type(16))) _Float16 v16h;
typedef __attribute__((ext_vector_type(8)))  _Float16 v8h;
typedef __attribute__((ext_vector_type(8)))  float    v8f;
typedef __attribute__((ext_vector_type(2)))  int      i32x2;
typedef __attribute__((ext_vector_type(4))) unsigned  tdm_u32x4;
typedef __attribute__((ext_vector_type(8))) int       tdm_i32x8;
typedef __attribute__((ext_vector_type(4))) int       tdm_i32x4;

// ---------------- CDNA5 feature probes ----------------
#if defined(__has_builtin)
#if __has_builtin(__builtin_amdgcn_global_load_async_to_lds_b64)
#define HAVE_ASYNC_LDS 1
#endif
#if __has_builtin(__builtin_amdgcn_tensor_load_to_lds)
#define HAVE_TDM 1
#endif
#if __has_builtin(__builtin_amdgcn_s_wait_asynccnt)
#define WAIT_ASYNC(n) __builtin_amdgcn_s_wait_asynccnt(n)
#endif
#endif
#ifndef HAVE_ASYNC_LDS
#define HAVE_ASYNC_LDS 0
#endif
#ifndef HAVE_TDM
#define HAVE_TDM 0
#endif
#ifndef WAIT_ASYNC
#define WAIT_ASYNC(n) asm volatile("s_wait_asynccnt " #n ::: "memory")
#endif

// Address-space bridging via integer casts (generic LDS ptr low 32 bits = LDS offset;
// generic global ptr == global address). Pointee type = <2 x i32> for the b64 builtin.
#define TO_GBL64(p) ((__attribute__((address_space(1))) i32x2*)(unsigned long long)(p))
#define TO_LDS64(p) ((__attribute__((address_space(3))) i32x2*)(unsigned)(unsigned long long)(p))

static __device__ __forceinline__ unsigned ldsOffset(const void* p) {
    return (unsigned)(unsigned long long)p;
}

// ---------------- device helpers ----------------
__device__ __forceinline__ float silu_f(float x) { return x / (1.f + __expf(-x)); }
__device__ __forceinline__ float softplus_f(float x) { return (x > 20.f) ? x : log1pf(__expf(x)); }

__device__ __forceinline__ float blockSum192(float v, float* s, int tid) {
    s[tid] = v;
    __syncthreads();
    if (tid < 64) s[tid] = s[tid] + s[tid + 64] + s[tid + 128];
    __syncthreads();
    for (int off = 32; off > 0; off >>= 1) {
        if (tid < off) s[tid] += s[tid + off];
        __syncthreads();
    }
    float total = s[0];
    __syncthreads();
    return total;
}

__device__ __forceinline__ float ln192(float x, float w, float b, float* s, int tid) {
    float sum = blockSum192(x, s, tid);
    float sq  = blockSum192(x * x, s, tid);
    float mean = sum * (1.f / 192.f);
    float var  = sq * (1.f / 192.f) - mean * mean;
    return (x - mean) * rsqrtf(var + LN_EPS) * w + b;
}

#if HAVE_TDM
// Issue one TDM 2D tile load: tileDim1 lines of tileDim0 2-byte elements,
// line stride strideElems (2-byte units), into contiguous LDS at lds_off.
__device__ __forceinline__ void tdm_load_2d(unsigned lds_off, unsigned long long ga,
                                            unsigned tileDim0, unsigned tileDim1,
                                            unsigned long long strideElems) {
    tdm_u32x4 g0;
    g0[0] = 1u;                                                    // count=1 (valid), user mode
    g0[1] = lds_off;                                               // lds_addr (bytes)
    g0[2] = (unsigned)(ga & 0xffffffffu);                          // global_addr lo
    g0[3] = (unsigned)((ga >> 32) & 0x01ffffffu) | (2u << 30);     // global_addr hi | type=2
    tdm_i32x8 g1;
    g1[0] = (int)(1u << 16);                                       // data_size = 2 bytes
    g1[1] = (int)((tileDim0 & 0xffffu) << 16);                     // tensor_dim0[15:0]
    g1[2] = (int)((tileDim0 >> 16) | ((tileDim1 & 0xffffu) << 16));// tensor_dim0 hi | tensor_dim1 lo
    g1[3] = (int)((tileDim1 >> 16) | (tileDim0 << 16));            // tensor_dim1 hi | tile_dim0
    g1[4] = (int)(tileDim1 & 0xffffu);                             // tile_dim1 (tile_dim2 = 0)
    g1[5] = (int)(unsigned)(strideElems & 0xffffffffu);            // tensor_dim0_stride lo32
    g1[6] = (int)(unsigned)((strideElems >> 32) & 0xffffu);        // tensor_dim0_stride hi16
    g1[7] = 0;
    tdm_i32x4 z4 = {0, 0, 0, 0};
    tdm_i32x8 z8 = {0, 0, 0, 0, 0, 0, 0, 0};
    __builtin_amdgcn_tensor_load_to_lds(g0, g1, z4, z4, z8, 0);    // 6-arg (clang-23) form
}
#endif

// ---------------- weight / activation packing ----------------
// src fp32 [NB][N][K] -> dst f16 [NB][K/32][N][32]  (WMMA B-fragment friendly)
__global__ void pack_weight_kernel(const float* __restrict__ src, _Float16* __restrict__ dst,
                                   int NB, int N, int K) {
    int idx = blockIdx.x * blockDim.x + threadIdx.x;
    int total = NB * N * K;
    if (idx >= total) return;
    int k = idx % K;
    int n = (idx / K) % N;
    int b = idx / (K * N);
    dst[(size_t)b * N * K + (size_t)(k >> 5) * N * 32 + (size_t)n * 32 + (k & 31)] =
        (_Float16)src[idx];
}

// Gather 16x16 patches of x (8,3,224,224) into row-major f16 A [1568][768]
__global__ void pack_patches_kernel(const float* __restrict__ x, _Float16* __restrict__ Ap) {
    int idx = blockIdx.x * blockDim.x + threadIdx.x;
    if (idx >= MROWS * KPATCH) return;
    int k = idx % KPATCH;
    int m = idx / KPATCH;
    int b = m / LSEQ, l = m % LSEQ;
    int hh = l / HGRID, ww = l % HGRID;
    int c = k / 256, py = (k % 256) / 16, px = k % 16;
    size_t off = (((size_t)(b * 3 + c) * 224) + (size_t)(hh * 16 + py)) * 224 + (ww * 16 + px);
    Ap[idx] = (_Float16)x[off];
}

// ---------------- WMMA GEMM with async-LDS / TDM staging ----------------
// Block = 8 waves -> 32(M) x 64(N) tile. A: f16 row-major [M][K] (async -> LDS).
// W: packed f16 [K/32][N][32] (TDM 2D tile -> LDS). Double-buffered K superchunks of 96.
template <int K, bool HAS_BIAS>
__global__ __launch_bounds__(256) void wmma_gemm_lds_kernel(
        const _Float16* __restrict__ Aall, size_t aStride,
        const _Float16* __restrict__ Wall, size_t wStride,
        const float* __restrict__ bias,
        float* __restrict__ Oall, size_t oStride, int N) {
    static_assert(K % KSUB == 0, "K must be a multiple of 96");
    constexpr int S = K / KSUB;
    __shared__ _Float16 As[2][32][AROWPAD];
    __shared__ _Float16 Bs[2][KSUB / 32][64][32];

    const int tid = threadIdx.x;
    const int lane = tid & 31;
    const int w = tid >> 5;
    const int wm = w >> 2;            // 0..1
    const int wn = w & 3;             // 0..3
    const int r = lane & 15;
    const int g = lane >> 4;
    const int mBase = blockIdx.x * 32;
    const int nBase = blockIdx.y * 64;
    const int bb = blockIdx.z;
    const _Float16* A = Aall + (size_t)bb * aStride;
    const _Float16* W = Wall + (size_t)bb * wStride;
    float* O = Oall + (size_t)bb * oStride;

    // stage A superchunk s into buffer pb: 32 rows x 96 halves, 3 b64 loads/thread
    auto stageA = [&](int s, int pb) {
#pragma unroll
        for (int i = 0; i < 3; ++i) {
            int j = tid + 256 * i;            // 0..767
            int row = j / 24;                 // 24 b64 per row
            int kp = (j % 24) * 4;            // half index within superchunk
            const _Float16* gp = A + (size_t)(mBase + row) * K + s * KSUB + kp;
            _Float16* lp = &As[pb][row][kp];
#if HAVE_ASYNC_LDS
            __builtin_amdgcn_global_load_async_to_lds_b64(TO_GBL64(gp), TO_LDS64(lp), 0, 0);
#else
            *(unsigned long long*)lp = *(const unsigned long long*)gp;
#endif
        }
    };
    // stage B superchunk s: 3 packed-fragment lines of 64*32 halves, line stride N*32
    auto stageB = [&](int s, int pb) {
#if HAVE_TDM
        if (tid == 0) {
            const _Float16* gp = W + ((size_t)(s * (KSUB / 32)) * N + nBase) * 32;
            tdm_load_2d(ldsOffset(&Bs[pb][0][0][0]), (unsigned long long)gp,
                        64 * 32, KSUB / 32, (unsigned long long)N * 32);
        }
#else
        _Float16* lbase = &Bs[pb][0][0][0];
        for (int j = tid; j < (KSUB / 32) * 64 * 32 / 4; j += 256) {
            int line = j / 512;
            int e = (j % 512) * 4;
            const _Float16* gp = W + ((size_t)(s * (KSUB / 32) + line) * N + nBase) * 32 + e;
            *(unsigned long long*)(lbase + line * 2048 + e) = *(const unsigned long long*)gp;
        }
#endif
    };

    stageA(0, 0);
    stageB(0, 0);

    const int aRow = wm * 16 + r;
    const int cLoc = wn * 16 + r;
    v8f acc = {};
    for (int s = 0; s < S; ++s) {
        const int pb = s & 1;
        __syncthreads();                       // all waves done with buffer pb^1
        if (s + 1 < S) {
            stageA(s + 1, pb ^ 1);
            stageB(s + 1, pb ^ 1);
            WAIT_ASYNC(3);                     // superchunk s A-data landed (s+1 in flight)
            __builtin_amdgcn_s_wait_tensorcnt(1);
        } else {
            WAIT_ASYNC(0);
            __builtin_amdgcn_s_wait_tensorcnt(0);
        }
        __syncthreads();                       // publish LDS writes to all waves
#pragma unroll
        for (int kc = 0; kc < KSUB / 32; ++kc) {
            v8h alo = *(const v8h*)&As[pb][aRow][kc * 32 + 16 * g];
            v8h ahi = *(const v8h*)&As[pb][aRow][kc * 32 + 16 * g + 8];
            v16h af;
#pragma unroll
            for (int i = 0; i < 8; ++i) { af[i] = alo[i]; af[i + 8] = ahi[i]; }
            v16h bf = *(const v16h*)&Bs[pb][kc][cLoc][16 * g];
            acc = __builtin_amdgcn_wmma_f32_16x16x32_f16(false, af, false, bf,
                                                         (short)0, acc, false, false);
        }
    }
    float bv = 0.f;
    if (HAS_BIAS) bv = bias[nBase + cLoc];
#pragma unroll
    for (int v = 0; v < 8; ++v) {
        int m = mBase + wm * 16 + 8 * g + v;   // C/D layout: rows 8g+v, col = lane%16
        O[(size_t)m * N + nBase + cLoc] = acc[v] + bv;
    }
}

// ---------------- LayerNorm over rows of 192 ----------------
__global__ void ln_rows_kernel(const float* __restrict__ in, const float* __restrict__ w,
                               const float* __restrict__ b, float* __restrict__ out) {
    __shared__ float s[192];
    int m = blockIdx.x, tid = threadIdx.x;
    float x = in[(size_t)m * DCH + tid];
    out[(size_t)m * DCH + tid] = ln192(x, w[tid], b[tid], s, tid);
}

// Build the 4 directional sequences; zero residual.
__global__ void init_dirs_kernel(const float* __restrict__ feat, float* __restrict__ hidden,
                                 float* __restrict__ residual) {
    int idx = blockIdx.x * blockDim.x + threadIdx.x;
    if (idx >= MG * DCH) return;
    int d = idx % DCH;
    int l = (idx / DCH) % LSEQ;
    int b = (idx / (DCH * LSEQ)) % BATCH;
    int g = idx / (DCH * LSEQ * BATCH);
    int lr = (g >= 2) ? (LSEQ - 1 - l) : l;                       // reversed dirs
    int lf = (g & 1) ? ((lr % HGRID) * HGRID + lr / HGRID) : lr;  // transposed dirs
    hidden[idx] = feat[((size_t)b * LSEQ + lf) * DCH + d];
    residual[idx] = 0.f;
}

// residual += hidden; xn16 = f16(LN(residual)).  grid=(1568, 4), block=192.
__global__ void preln_kernel(float* __restrict__ residual, const float* __restrict__ hidden,
                             const float* __restrict__ ln_w, const float* __restrict__ ln_b,
                             _Float16* __restrict__ xn16, int layer) {
    __shared__ float s[192];
    int m = blockIdx.x, g = blockIdx.y, tid = threadIdx.x;
    size_t idx = ((size_t)g * MROWS + m) * DCH + tid;
    float r = residual[idx] + hidden[idx];
    residual[idx] = r;
    int wi = (g * DEPTH + layer) * DCH + tid;
    xn16[idx] = (_Float16)ln192(r, ln_w[wi], ln_b[wi], s, tid);
}

// causal depthwise conv(4) + bias + SiLU over u = xz[..., :384]
__global__ void conv_silu_kernel(const float* __restrict__ xz, const float* __restrict__ conv_w,
                                 const float* __restrict__ conv_b, float* __restrict__ u,
                                 _Float16* __restrict__ u16, int layer) {
    int idx = blockIdx.x * blockDim.x + threadIdx.x;
    if (idx >= MG * DIN) return;
    int d = idx % DIN;
    int l = (idx / DIN) % LSEQ;
    int bg = idx / (DIN * LSEQ);   // g*BATCH + b
    int g = bg / BATCH;
    size_t seqbase = (size_t)bg * LSEQ;
    const float* cw = conv_w + ((size_t)(g * DEPTH + layer) * DIN + d) * DCONV;
    float acc = conv_b[(g * DEPTH + layer) * DIN + d];
#pragma unroll
    for (int j = 0; j < DCONV; ++j) {
        int ll = l - (DCONV - 1) + j;
        if (ll >= 0) acc += xz[(seqbase + ll) * XZW + d] * cw[j];
    }
    float su = silu_f(acc);
    u[idx] = su;
    u16[idx] = (_Float16)su;
}

// xdbl = u @ x_proj_w^T (44 outs), delta = softplus(dt @ dt_w^T + dt_b).
__global__ void xproj_delta_kernel(const float* __restrict__ u, const float* __restrict__ xpw,
                                   const float* __restrict__ dt_w, const float* __restrict__ dt_b,
                                   float* __restrict__ xdbl, float* __restrict__ delta,
                                   int layer) {
    __shared__ float uL[DIN];
    __shared__ float xd[DTR + 2 * DSTATE];   // 44
    int m = blockIdx.x, g = blockIdx.y, tid = threadIdx.x;
    size_t mb = (size_t)g * MROWS + m;
    for (int i = tid; i < DIN; i += blockDim.x) uL[i] = u[mb * DIN + i];
    __syncthreads();
    if (tid < DTR + 2 * DSTATE) {
        const float* w = xpw + ((size_t)(g * DEPTH + layer) * (DTR + 2 * DSTATE) + tid) * DIN;
        float sacc = 0.f;
        for (int k = 0; k < DIN; ++k) sacc += uL[k] * w[k];
        xd[tid] = sacc;
        xdbl[mb * (DTR + 2 * DSTATE) + tid] = sacc;
    }
    __syncthreads();
    for (int d = tid; d < DIN; d += blockDim.x) {
        const float* dw = dt_w + ((size_t)(g * DEPTH + layer) * DIN + d) * DTR;
        float sacc = dt_b[(g * DEPTH + layer) * DIN + d];
#pragma unroll
        for (int rr = 0; rr < DTR; ++rr) sacc += xd[rr] * dw[rr];
        delta[mb * DIN + d] = softplus_f(sacc);
    }
}

// Selective scan. grid=(8, 4), block=384 (one lane per channel), state in registers.
__global__ void scan_kernel(const float* __restrict__ delta, const float* __restrict__ u,
                            const float* __restrict__ xdbl, const float* __restrict__ xz,
                            const float* __restrict__ A_log, const float* __restrict__ Dp,
                            _Float16* __restrict__ y16, int layer) {
    int b = blockIdx.x, g = blockIdx.y, d = threadIdx.x;
    const float* al = A_log + ((size_t)(g * DEPTH + layer) * DIN + d) * DSTATE;
    float Ar[DSTATE];
#pragma unroll
    for (int n = 0; n < DSTATE; ++n) Ar[n] = -__expf(al[n]);
    float Dd = Dp[(g * DEPTH + layer) * DIN + d];
    float h[DSTATE];
#pragma unroll
    for (int n = 0; n < DSTATE; ++n) h[n] = 0.f;
    size_t base = (size_t)g * MROWS + (size_t)b * LSEQ;
    for (int l = 0; l < LSEQ; ++l) {
        size_t m = base + l;
        float dl = delta[m * DIN + d];
        float ud = u[m * DIN + d];
        const float* xdb = xdbl + m * (DTR + 2 * DSTATE);
        float du = dl * ud;
        float y = 0.f;
#pragma unroll
        for (int n = 0; n < DSTATE; ++n) {
            h[n] = h[n] * __expf(dl * Ar[n]) + du * xdb[DTR + n];
            y += h[n] * xdb[DTR + DSTATE + n];
        }
        y += ud * Dd;
        float z = xz[m * XZW + DIN + d];
        y *= silu_f(z);
        y16[m * DIN + d] = (_Float16)y;
    }
}

// Merge the 4 direction outputs (hidden + residual) spatially.
__global__ void combine_kernel(const float* __restrict__ hidden,
                               const float* __restrict__ residual,
                               float* __restrict__ ycomb) {
    int idx = blockIdx.x * blockDim.x + threadIdx.x;
    if (idx >= BATCH * LSEQ * DCH) return;
    int d = idx % DCH;
    int l = (idx / DCH) % LSEQ;
    int b = idx / (DCH * LSEQ);
    int l1 = (l % HGRID) * HGRID + l / HGRID;   // transposed position
    auto ov = [&](int g, int ls) -> float {
        size_t j = ((size_t)g * MROWS + (size_t)b * LSEQ + ls) * DCH + d;
        return hidden[j] + residual[j];
    };
    ycomb[idx] = ov(0, l) + ov(2, LSEQ - 1 - l) + ov(1, l1) + ov(3, LSEQ - 1 - l1);
}

// Two stacked LayerNorms (out_norm, head_ln). grid=1568, block=192.
__global__ void final_ln_kernel(const float* __restrict__ ycomb,
                                const float* __restrict__ onw, const float* __restrict__ onb,
                                const float* __restrict__ hlw, const float* __restrict__ hlb,
                                float* __restrict__ yln) {
    __shared__ float s[192];
    int m = blockIdx.x, tid = threadIdx.x;
    float x = ycomb[(size_t)m * DCH + tid];
    x = ln192(x, onw[tid], onb[tid], s, tid);
    x = ln192(x, hlw[tid], hlb[tid], s, tid);
    yln[(size_t)m * DCH + tid] = x;
}

// Deterministic mean pooling over L (no float atomics -> bitwise-stable replays).
__global__ void pool_kernel(const float* __restrict__ yln, float* __restrict__ pooled) {
    int idx = blockIdx.x * blockDim.x + threadIdx.x;
    if (idx >= BATCH * DCH) return;
    int d = idx % DCH, b = idx / DCH;
    float s = 0.f;
    for (int l = 0; l < LSEQ; ++l) s += yln[((size_t)b * LSEQ + l) * DCH + d];
    pooled[idx] = s * (1.f / (float)LSEQ);
}

__global__ void head_kernel(const float* __restrict__ pooled, const float* __restrict__ hw,
                            const float* __restrict__ hb, float* __restrict__ out) {
    int idx = blockIdx.x * blockDim.x + threadIdx.x;
    if (idx >= BATCH * NCLS) return;
    int c = idx % NCLS, b = idx / NCLS;
    const float* p = pooled + b * DCH;
    const float* w = hw + (size_t)c * DCH;
    float sacc = hb[c];
    for (int d = 0; d < DCH; ++d) sacc += p[d] * w[d];
    out[idx] = sacc;
}

// ---------------- host-side orchestration ----------------
extern "C" void kernel_launch(void* const* d_in, const int* in_sizes, int n_in,
                              void* d_out, int out_size, void* d_ws, size_t ws_size,
                              hipStream_t stream) {
    (void)in_sizes; (void)n_in; (void)out_size; (void)ws_size;
    const float* x          = (const float*)d_in[0];
    const float* patch_w    = (const float*)d_in[1];
    const float* patch_b    = (const float*)d_in[2];
    const float* pe_ln_w    = (const float*)d_in[3];
    const float* pe_ln_b    = (const float*)d_in[4];
    const float* ln_w       = (const float*)d_in[5];
    const float* ln_b       = (const float*)d_in[6];
    const float* in_proj_w  = (const float*)d_in[7];
    const float* conv_w     = (const float*)d_in[8];
    const float* conv_b     = (const float*)d_in[9];
    const float* x_proj_w   = (const float*)d_in[10];
    const float* dt_w       = (const float*)d_in[11];
    const float* dt_b       = (const float*)d_in[12];
    const float* A_log      = (const float*)d_in[13];
    const float* Dp         = (const float*)d_in[14];
    const float* out_proj_w = (const float*)d_in[15];
    const float* out_norm_w = (const float*)d_in[16];
    const float* out_norm_b = (const float*)d_in[17];
    const float* head_ln_w  = (const float*)d_in[18];
    const float* head_ln_b  = (const float*)d_in[19];
    const float* head_w     = (const float*)d_in[20];
    const float* head_b     = (const float*)d_in[21];
    float* out = (float*)d_out;

    size_t off = 0;
    auto alloc = [&](size_t bytes) -> void* {
        size_t o = (off + 255) & ~(size_t)255;
        off = o + bytes;
        return (void*)((char*)d_ws + o);
    };
    _Float16* Apatch  = (_Float16*)alloc((size_t)MROWS * KPATCH * 2);
    _Float16* WpPatch = (_Float16*)alloc((size_t)KPATCH * DCH * 2);
    _Float16* WpIn    = (_Float16*)alloc((size_t)DIRS * DEPTH * XZW * DCH * 2);
    _Float16* WpOut   = (_Float16*)alloc((size_t)DIRS * DEPTH * DCH * DIN * 2);
    float*    featraw = (float*)alloc((size_t)MROWS * DCH * 4);
    float*    feat    = (float*)alloc((size_t)MROWS * DCH * 4);
    float*    hidden  = (float*)alloc((size_t)MG * DCH * 4);
    float*    residual= (float*)alloc((size_t)MG * DCH * 4);
    _Float16* xn16    = (_Float16*)alloc((size_t)MG * DCH * 2);
    float*    xz      = (float*)alloc((size_t)MG * XZW * 4);
    float*    u       = (float*)alloc((size_t)MG * DIN * 4);
    _Float16* u16     = (_Float16*)alloc((size_t)MG * DIN * 2);
    float*    xdbl    = (float*)alloc((size_t)MG * (DTR + 2 * DSTATE) * 4);
    float*    delta   = (float*)alloc((size_t)MG * DIN * 4);
    _Float16* y16     = (_Float16*)alloc((size_t)MG * DIN * 2);
    float*    ycomb   = (float*)alloc((size_t)MROWS * DCH * 4);
    float*    yln     = (float*)alloc((size_t)MROWS * DCH * 4);
    float*    pooled  = (float*)alloc((size_t)BATCH * DCH * 4);

    const int TPB = 256;
    auto nb = [](size_t n, int t) { return (unsigned)((n + t - 1) / t); };

    // ---- weight packing (f16, WMMA-fragment layout) ----
    pack_weight_kernel<<<nb((size_t)DCH * KPATCH, TPB), TPB, 0, stream>>>(
        patch_w, WpPatch, 1, DCH, KPATCH);
    pack_weight_kernel<<<nb((size_t)32 * XZW * DCH, TPB), TPB, 0, stream>>>(
        in_proj_w, WpIn, DIRS * DEPTH, XZW, DCH);
    pack_weight_kernel<<<nb((size_t)32 * DCH * DIN, TPB), TPB, 0, stream>>>(
        out_proj_w, WpOut, DIRS * DEPTH, DCH, DIN);

    // ---- patch embedding: gather -> WMMA GEMM (+bias) -> LN ----
    pack_patches_kernel<<<nb((size_t)MROWS * KPATCH, TPB), TPB, 0, stream>>>(x, Apatch);
    wmma_gemm_lds_kernel<KPATCH, true><<<dim3(MROWS / 32, DCH / 64, 1), 256, 0, stream>>>(
        Apatch, 0, WpPatch, 0, patch_b, featraw, 0, DCH);
    ln_rows_kernel<<<MROWS, 192, 0, stream>>>(featraw, pe_ln_w, pe_ln_b, feat);
    init_dirs_kernel<<<nb((size_t)MG * DCH, TPB), TPB, 0, stream>>>(feat, hidden, residual);

    // ---- 8 Mamba layers, 4 directions batched in grid.z ----
    const size_t chunkIn  = (size_t)XZW * DCH;   // packed halves per in_proj matrix
    const size_t chunkOut = (size_t)DCH * DIN;   // packed halves per out_proj matrix
    for (int layer = 0; layer < DEPTH; ++layer) {
        preln_kernel<<<dim3(MROWS, DIRS), 192, 0, stream>>>(
            residual, hidden, ln_w, ln_b, xn16, layer);
        wmma_gemm_lds_kernel<DCH, false><<<dim3(MROWS / 32, XZW / 64, DIRS), 256, 0, stream>>>(
            xn16, (size_t)MROWS * DCH, WpIn + (size_t)layer * chunkIn,
            (size_t)DEPTH * chunkIn, nullptr, xz, (size_t)MROWS * XZW, XZW);
        conv_silu_kernel<<<nb((size_t)MG * DIN, TPB), TPB, 0, stream>>>(
            xz, conv_w, conv_b, u, u16, layer);
        xproj_delta_kernel<<<dim3(MROWS, DIRS), 128, 0, stream>>>(
            u, x_proj_w, dt_w, dt_b, xdbl, delta, layer);
        scan_kernel<<<dim3(BATCH, DIRS), DIN, 0, stream>>>(
            delta, u, xdbl, xz, A_log, Dp, y16, layer);
        wmma_gemm_lds_kernel<DIN, false><<<dim3(MROWS / 32, DCH / 64, DIRS), 256, 0, stream>>>(
            y16, (size_t)MROWS * DIN, WpOut + (size_t)layer * chunkOut,
            (size_t)DEPTH * chunkOut, nullptr, hidden, (size_t)MROWS * DCH, DCH);
    }

    // ---- merge directions, double LN, pool, head ----
    combine_kernel<<<nb((size_t)MROWS * DCH, TPB), TPB, 0, stream>>>(hidden, residual, ycomb);
    final_ln_kernel<<<MROWS, 192, 0, stream>>>(
        ycomb, out_norm_w, out_norm_b, head_ln_w, head_ln_b, yln);
    pool_kernel<<<nb((size_t)BATCH * DCH, TPB), TPB, 0, stream>>>(yln, pooled);
    head_kernel<<<nb((size_t)BATCH * NCLS, TPB), TPB, 0, stream>>>(pooled, head_w, head_b, out);
}
